// MixtralMoE_41686952575380
// MI455X (gfx1250) — matrix operand order repaired and checked by reading, not compiled, not run on hardware
//
#include <hip/hip_runtime.h>
#include <hip/hip_bf16.h>

// Mixtral MoE for MI455X (gfx1250): bf16 WMMA (v_wmma_f32_16x16x32_bf16),
// top-2 dispatch + combine, fp32 accumulate. LDS tiles in bf16 so each lane's
// WMMA fragment is two aligned ds_load_b128's; fp32->bf16 conversion fused
// into weight staging; activation tile in GEMM2 streamed with
// global_load_async_to_lds_b128 (ASYNCcnt) since it needs no conversion.

static constexpr int Hd = 1024;   // hidden
static constexpr int Id = 3584;   // intermediate
static constexpr int Td = 2048;   // tokens (2*1024)
static constexpr int Ed = 8;      // experts

typedef __attribute__((ext_vector_type(16))) __bf16 v16bf;
typedef __attribute__((ext_vector_type(8)))  __bf16 v8bf;
typedef __attribute__((ext_vector_type(4)))  __bf16 v4bf;
typedef __attribute__((ext_vector_type(8)))  float  v8f;

static constexpr int LDA = 40;    // bf16 tile row stride (80B: keeps 16B align)

__device__ __forceinline__ v4bf cvt4(float4 v) {
  v4bf r;
  r[0] = (__bf16)v.x; r[1] = (__bf16)v.y;
  r[2] = (__bf16)v.z; r[3] = (__bf16)v.w;
  return r;
}

// LDS byte offset of a __shared__ object (AS3 pointers are 32-bit LDS offsets).
__device__ __forceinline__ unsigned lds_off_u32(const void* p) {
  return (unsigned)(unsigned long long)(__attribute__((address_space(3))) const void*)p;
}

// A (16x32 MxK) row-major bf16 tile (stride LDA):
// lane 0-15 -> M=lane, K={0..7,16..23}; lane 16-31 -> M=lane-16, K={8..15,24..31}
__device__ __forceinline__ v16bf pack_a(const __bf16* __restrict__ T, int lane) {
  const int m  = lane & 15;
  const int kb = (lane & 16) ? 8 : 0;
  const v8bf lo = *(const v8bf*)(T + m * LDA + kb);        // K = kb..kb+7
  const v8bf hi = *(const v8bf*)(T + m * LDA + 16 + kb);   // K = 16+kb..16+kb+7
  return __builtin_shufflevector(lo, hi, 0, 1, 2, 3, 4, 5, 6, 7,
                                 8, 9, 10, 11, 12, 13, 14, 15);
}
// B (32x16 KxN) from n-major bf16 tile [128][LDA] (row n = W row over K):
// lane 0-15 -> N=lane, K={0..7,16..23}; lane 16-31 -> N=lane-16, K={8..15,24..31}
__device__ __forceinline__ v16bf pack_b(const __bf16* __restrict__ T, int lane, int ncol) {
  const int n  = ncol + (lane & 15);
  const int kb = (lane & 16) ? 8 : 0;
  const v8bf lo = *(const v8bf*)(T + n * LDA + kb);
  const v8bf hi = *(const v8bf*)(T + n * LDA + 16 + kb);
  return __builtin_shufflevector(lo, hi, 0, 1, 2, 3, 4, 5, 6, 7,
                                 8, 9, 10, 11, 12, 13, 14, 15);
}

// ---- Router: logits -> softmax -> top2 -> renorm -> dispatch lists -----
__global__ __launch_bounds__(256) void moe_router(
    const float* __restrict__ x, const float* __restrict__ wg,
    int* __restrict__ cnt, int* __restrict__ list, float* __restrict__ wgt) {
  const int t    = blockIdx.x * 8 + (threadIdx.x >> 5);
  const int lane = threadIdx.x & 31;
  float acc[Ed];
#pragma unroll
  for (int e = 0; e < Ed; ++e) acc[e] = 0.0f;
  for (int h = lane; h < Hd; h += 32) {
    const float xv = x[(size_t)t * Hd + h];
#pragma unroll
    for (int e = 0; e < Ed; ++e) acc[e] += xv * wg[e * Hd + h];
  }
#pragma unroll
  for (int off = 16; off > 0; off >>= 1) {
#pragma unroll
    for (int e = 0; e < Ed; ++e) acc[e] += __shfl_xor(acc[e], off, 32);
  }
  if (lane == 0) {
    float mx = acc[0];
#pragma unroll
    for (int e = 1; e < Ed; ++e) mx = fmaxf(mx, acc[e]);
    float p[Ed], s = 0.0f;
#pragma unroll
    for (int e = 0; e < Ed; ++e) { p[e] = __expf(acc[e] - mx); s += p[e]; }
#pragma unroll
    for (int e = 0; e < Ed; ++e) p[e] /= s;
    int i1 = 0; float p1 = p[0];
#pragma unroll
    for (int e = 1; e < Ed; ++e) if (p[e] > p1) { p1 = p[e]; i1 = e; }
    int i2 = -1; float p2 = -1.0f;
#pragma unroll
    for (int e = 0; e < Ed; ++e) if (e != i1 && p[e] > p2) { p2 = p[e]; i2 = e; }
    const float den = p1 + p2;
    int pos = atomicAdd(&cnt[i1], 1);
    list[i1 * Td + pos] = t; wgt[i1 * Td + pos] = p1 / den;
    pos = atomicAdd(&cnt[i2], 1);
    list[i2 * Td + pos] = t; wgt[i2 * Td + pos] = p2 / den;
  }
}

// ---- GEMM1: h = silu(Xg @ W1^T) * (Xg @ W3^T) for expert e -------------
// grid (I/128, T/32), block 256 = 8 waves; wave w owns N-subtile w*16,
// two 16-row A tiles per block (M=32).
__global__ __launch_bounds__(256) void moe_gemm1(
    const float* __restrict__ x, const float* __restrict__ w1,
    const float* __restrict__ w3, const int* __restrict__ cnt,
    const int* __restrict__ list, __bf16* __restrict__ hbuf, int e) {
  __shared__ __align__(16) __bf16 Xt[32 * LDA];
  __shared__ __align__(16) __bf16 W1t[128 * LDA];
  __shared__ __align__(16) __bf16 W3t[128 * LDA];

  const int ne = cnt[e];
  const int m0 = blockIdx.y * 32;
  if (m0 >= ne) return;                       // block-uniform: EXEC stays full
  const int n0   = blockIdx.x * 128;
  const int tid  = threadIdx.x;
  const int lane = tid & 31;
  const int wv   = tid >> 5;

  const float* w1e = w1 + (size_t)e * Id * Hd;
  const float* w3e = w3 + (size_t)e * Id * Hd;

  v8f g0 = {}, g1 = {};
  v8f u0 = {}, u1 = {};

  for (int k0 = 0; k0 < Hd; k0 += 32) {
    {                                         // X tile: 32 rows x 32 (gather)
      const int r = tid >> 3;                 // 0..31
      const int c = (tid & 7) * 4;
      const int p = m0 + r;
      float4 v = make_float4(0.f, 0.f, 0.f, 0.f);
      if (p < ne) {
        const int t = list[e * Td + p];
        v = *(const float4*)(x + (size_t)t * Hd + k0 + c);
      }
      *(v4bf*)(&Xt[r * LDA + c]) = cvt4(v);
    }
#pragma unroll
    for (int q = 0; q < 4; ++q) {             // W tiles: 128 rows x 32
      const int f = tid + q * 256;
      const int r = f >> 3;
      const int c = (f & 7) * 4;
      const size_t off = (size_t)(n0 + r) * Hd + (k0 + c);
      *(v4bf*)(&W1t[r * LDA + c]) = cvt4(*(const float4*)(w1e + off));
      *(v4bf*)(&W3t[r * LDA + c]) = cvt4(*(const float4*)(w3e + off));
      if (k0 + 32 < Hd) {                     // prefetch next k-step weight lines
        __builtin_prefetch(w1e + off + 32, 0, 1);
        __builtin_prefetch(w3e + off + 32, 0, 1);
      }
    }
    __syncthreads();
    const v16bf a0 = pack_a(Xt, lane);
    const v16bf a1 = pack_a(Xt + 16 * LDA, lane);
    const v16bf b1 = pack_b(W1t, lane, wv * 16);
    const v16bf b3 = pack_b(W3t, lane, wv * 16);
    g0 = __builtin_amdgcn_wmma_f32_16x16x32_bf16(false, a0, false, b1, (short)0, g0, false, false);
    g1 = __builtin_amdgcn_wmma_f32_16x16x32_bf16(false, a1, false, b1, (short)0, g1, false, false);
    u0 = __builtin_amdgcn_wmma_f32_16x16x32_bf16(false, a0, false, b3, (short)0, u0, false, false);
    u1 = __builtin_amdgcn_wmma_f32_16x16x32_bf16(false, a1, false, b3, (short)0, u1, false, false);
    __syncthreads();
  }

  const int col = n0 + wv * 16 + (lane & 15); // C/D layout: vgpr r -> M, lane -> N
#pragma unroll
  for (int r = 0; r < 8; ++r) {
    const int rr   = (lane & 16) ? (8 + r) : r;
    const int row0 = m0 + rr;
    const int row1 = m0 + 16 + rr;
    if (row0 < ne) {
      const float gg = g0[r];
      const float hv = (gg / (1.0f + __expf(-gg))) * u0[r];
      hbuf[(size_t)row0 * Id + col] = (__bf16)hv;
    }
    if (row1 < ne) {
      const float gg = g1[r];
      const float hv = (gg / (1.0f + __expf(-gg))) * u1[r];
      hbuf[(size_t)row1 * Id + col] = (__bf16)hv;
    }
  }
}

// ---- GEMM2: y[t,:] += wgt * (h @ W2^T) for expert e --------------------
// grid (H/128, T/32), block 256 = 8 waves, M=32.
__global__ __launch_bounds__(256) void moe_gemm2(
    const float* __restrict__ w2, const __bf16* __restrict__ hbuf,
    const int* __restrict__ cnt, const int* __restrict__ list,
    const float* __restrict__ wgt, float* __restrict__ y, int e) {
  __shared__ __align__(16) __bf16 Ht[32 * LDA];
  __shared__ __align__(16) __bf16 W2t[128 * LDA];

  const int ne = cnt[e];
  const int m0 = blockIdx.y * 32;
  if (m0 >= ne) return;
  const int n0   = blockIdx.x * 128;
  const int tid  = threadIdx.x;
  const int lane = tid & 31;
  const int wv   = tid >> 5;

  const float* w2e = w2 + (size_t)e * Hd * Id;
  v8f acc0 = {}, acc1 = {};

  for (int k0 = 0; k0 < Id; k0 += 32) {
    // H tile: 32 rows x 32 bf16 = pure copy -> async global->LDS (ASYNCcnt).
    // Rows >= ne clamp to a valid row: their garbage only reaches output rows
    // the guarded epilogue never stores.
    if (tid < 128) {
      const int r  = tid >> 2;                // 0..31
      const int c  = (tid & 3) * 8;           // 0,8,16,24 (bf16 elems; 16B chunks)
      const int p  = m0 + r;
      const int pc = (p < ne) ? p : (ne - 1);
      const unsigned dst = lds_off_u32(&Ht[r * LDA + c]);
      const __bf16* src = hbuf + (size_t)pc * Id + k0 + c;
      asm volatile("global_load_async_to_lds_b128 %0, %1, off"
                   :: "v"(dst), "v"(src) : "memory");
    }
#pragma unroll
    for (int q = 0; q < 4; ++q) {             // W2 tile: 128 rows x 32
      const int f = tid + q * 256;
      const int r = f >> 3;
      const int c = (f & 7) * 4;
      const size_t off = (size_t)(n0 + r) * Id + (k0 + c);
      *(v4bf*)(&W2t[r * LDA + c]) = cvt4(*(const float4*)(w2e + off));
      if (k0 + 32 < Id) {                     // prefetch next k-step weight lines
        __builtin_prefetch(w2e + off + 32, 0, 1);
      }
    }
    asm volatile("s_wait_asynccnt 0" ::: "memory");
    __syncthreads();
    const v16bf a0 = pack_a(Ht, lane);
    const v16bf a1 = pack_a(Ht + 16 * LDA, lane);
    const v16bf b  = pack_b(W2t, lane, wv * 16);
    acc0 = __builtin_amdgcn_wmma_f32_16x16x32_bf16(false, a0, false, b, (short)0, acc0, false, false);
    acc1 = __builtin_amdgcn_wmma_f32_16x16x32_bf16(false, a1, false, b, (short)0, acc1, false, false);
    __syncthreads();
  }

  const int col = n0 + wv * 16 + (lane & 15);
#pragma unroll
  for (int r = 0; r < 8; ++r) {
    const int rr   = (lane & 16) ? (8 + r) : r;
    const int row0 = m0 + rr;
    const int row1 = m0 + 16 + rr;
    if (row0 < ne) {
      const int   t = list[e * Td + row0];
      const float w = wgt[e * Td + row0];
      atomicAdd(&y[(size_t)t * Hd + col], acc0[r] * w);
    }
    if (row1 < ne) {
      const int   t = list[e * Td + row1];
      const float w = wgt[e * Td + row1];
      atomicAdd(&y[(size_t)t * Hd + col], acc1[r] * w);
    }
  }
}

extern "C" void kernel_launch(void* const* d_in, const int* in_sizes, int n_in,
                              void* d_out, int out_size, void* d_ws, size_t ws_size,
                              hipStream_t stream) {
  (void)in_sizes; (void)n_in; (void)out_size; (void)ws_size;
  const float* x  = (const float*)d_in[0];   // [2,1024,H] fp32
  const float* wg = (const float*)d_in[1];   // [E,H]
  const float* w1 = (const float*)d_in[2];   // [E,I,H]
  const float* w3 = (const float*)d_in[3];   // [E,I,H]
  const float* w2 = (const float*)d_in[4];   // [E,H,I]
  float* y = (float*)d_out;                  // [2,1024,H] fp32

  // workspace layout
  char* ws = (char*)d_ws;
  int*    cnt  = (int*)ws;                                  // 8 ints (zeroed)
  int*    list = (int*)(ws + 64);                           // E*T ints
  float*  wgt  = (float*)(ws + 64 + (size_t)Ed * Td * 4);   // E*T f32
  __bf16* hbuf = (__bf16*)(ws + 64 + 2 * (size_t)Ed * Td * 4); // T*I bf16

  hipMemsetAsync(cnt, 0, 64, stream);
  hipMemsetAsync(y, 0, (size_t)Td * Hd * sizeof(float), stream);

  moe_router<<<Td / 8, 256, 0, stream>>>(x, wg, cnt, list, wgt);

  for (int e = 0; e < Ed; ++e) {             // stream order serializes deps
    moe_gemm1<<<dim3(Id / 128, Td / 32), 256, 0, stream>>>(x, w1, w3, cnt, list, hbuf, e);
    moe_gemm2<<<dim3(Hd / 128, Td / 32), 256, 0, stream>>>(w2, hbuf, cnt, list, wgt, y, e);
  }
}
